// LiquidBlock_21689584845455
// MI455X (gfx1250) — matrix operand order repaired and checked by reading, not compiled, not run on hardware
//
#include <hip/hip_runtime.h>

typedef __attribute__((ext_vector_type(16))) _Float16 v16h;
typedef __attribute__((ext_vector_type(8)))  _Float16 v8h;
typedef __attribute__((ext_vector_type(8)))  float    v8f;
typedef __attribute__((ext_vector_type(4)))  float    v4f;

namespace {
constexpr int kB = 128, kT = 4096, kF = 64;
constexpr int kH0 = 77, kH1 = 51, kH2 = 128, kUnits = 256;
// WMMA tiling: K padded to multiples of 32, N to multiples of 16.
constexpr int KT0 = 5, NT0 = 5;   // L0: k=141->160, h=77->80
constexpr int KT1 = 4, NT1 = 4;   // L1: k=128,      h=51->64
constexpr int KT2 = 6, NT2 = 8;   // L2: k=179->192, h=128
constexpr int FRAG = 512;         // f16 elems per 32x16 B-fragment (32 lanes x 16)
constexpr int F0 = 4 * NT0 * KT0; // 100 fragments
constexpr int F1 = 4 * NT1 * KT1; // 64
constexpr int F2 = 4 * NT2 * KT2; // 192
constexpr int BASE0 = 0, BASE1 = F0, BASE2 = F0 + F1;
constexpr int NFRAG = F0 + F1 + F2; // 356 -> 364544 bytes of f16 in d_ws
}

// ---------------- prep kernel: mask + f32->f16 + pack into B-fragment layout --
struct PrepParams {
  const float* mask[3];
  const float* W[3][4];
  _Float16* ws;
};

__global__ __launch_bounds__(512) void cfc_prep_weights(PrepParams p) {
  const int bid = blockIdx.x; // one block per fragment
  int l, r, KT, NT, h, k;
  if (bid < BASE1)      { l = 0; r = bid;         KT = KT0; NT = NT0; h = kH0; k = kF  + kH0; }
  else if (bid < BASE2) { l = 1; r = bid - BASE1; KT = KT1; NT = NT1; h = kH1; k = kH0 + kH1; }
  else                  { l = 2; r = bid - BASE2; KT = KT2; NT = NT2; h = kH2; k = kH1 + kH2; }
  const int kt = r % KT, nt = (r / KT) % NT, g = r / (KT * NT);
  const int tid = threadIdx.x;        // 0..511 = (lane, e)
  const int lane = tid >> 4, e = tid & 15;
  const int n = lane & 15, kb = (lane >> 4) << 4;   // B layout: lanes 0-15 K=0..15, 16-31 K=16..31
  const int row = nt * 16 + n;        // output neuron
  const int col = kt * 32 + kb + e;   // k index
  float v = 0.f;
  if (row < h && col < k) {
    const size_t idx = (size_t)row * k + col;
    v = p.W[l][g][idx] * p.mask[l][idx];
  }
  p.ws[(size_t)bid * FRAG + tid] = (_Float16)v;
}

// ---------------- persistent CfC kernel ---------------------------------------
struct MainParams {
  const float* x;            // (B,T,F)
  const float* h0;           // (B,UNITS)
  const float* bias[3][4];
  const _Float16* ws;
  float* out;                // (B,T,128)
  float* hx;                 // (B,UNITS) final state
};

struct SharedMem {
  // z buffers: [ping-pong][16 rows][K-padded], f16. Hidden parts double-buffered.
  alignas(16) _Float16 z0[2][16][160];  // [x(64) | h0(77) | pad]
  alignas(16) _Float16 z1[2][16][128];  // [h0(77) | h1(51)]
  alignas(16) _Float16 z2[2][16][192];  // [h1(51) | h2(128) | pad]
  float bias[3][4][128];
};

template <int KTN, typename Writer>
__device__ __forceinline__ void layer_compute(
    const _Float16* __restrict__ zbuf, int zstride,
    const _Float16* __restrict__ wslayer, int gate_stride,
    int nt, int lane, const float* __restrict__ bias, int hsz, Writer&& wr) {
  const int mA = lane & 15, half = lane >> 4;
  const _Float16* zrow = zbuf + mA * zstride;
  v8f a0 = {}, a1 = {}, a2 = {}, a3 = {};
  const _Float16* fb = wslayer + (size_t)nt * KTN * FRAG + lane * 16;
#pragma unroll
  for (int kt = 0; kt < KTN; ++kt) {
    // 16-bit A layout: lane holds K = ktb + half*16 + 0..15 (matches packed B)
    union { v16h v; v8h p[2]; } ua;
    ua.p[0] = *(const v8h*)(zrow + kt * 32 + half * 16);
    ua.p[1] = *(const v8h*)(zrow + kt * 32 + half * 16 + 8);
    v16h b0 = *(const v16h*)(fb);
    v16h b1 = *(const v16h*)(fb + gate_stride);
    v16h b2 = *(const v16h*)(fb + 2 * gate_stride);
    v16h b3 = *(const v16h*)(fb + 3 * gate_stride);
    a0 = __builtin_amdgcn_wmma_f32_16x16x32_f16(false, ua.v, false, b0, (short)0, a0, false, false);
    a1 = __builtin_amdgcn_wmma_f32_16x16x32_f16(false, ua.v, false, b1, (short)0, a1, false, false);
    a2 = __builtin_amdgcn_wmma_f32_16x16x32_f16(false, ua.v, false, b2, (short)0, a2, false, false);
    a3 = __builtin_amdgcn_wmma_f32_16x16x32_f16(false, ua.v, false, b3, (short)0, a3, false, false);
    fb += FRAG;
  }
  const int n = nt * 16 + mA;
  if (n < hsz) {
    const float b1v = bias[n], b2v = bias[128 + n], bav = bias[256 + n], bbv = bias[384 + n];
#pragma unroll
    for (int r2 = 0; r2 < 8; ++r2) {
      // C layout: VGPR r2 -> M = r2 + 8*half, N = lane%16
      const float f1 = tanhf(a0[r2] + b1v);
      const float f2 = tanhf(a1[r2] + b2v);
      const float s  = 1.f / (1.f + __expf(-(a2[r2] + bav + a3[r2] + bbv)));
      wr(r2 + 8 * half, n, f1 + s * (f2 - f1));
    }
  }
}

__global__ __launch_bounds__(256, 1) void cfc_persistent(MainParams p) {
  __shared__ SharedMem sh;
  const int tid = threadIdx.x, wave = tid >> 5, lane = tid & 31;
  const int brow = blockIdx.x * 16;   // this WG owns batch rows brow..brow+15

  { // zero all z buffers (pads must stay zero forever)
    unsigned* w = (unsigned*)&sh;
    constexpr int zw = (int)((sizeof(sh.z0) + sizeof(sh.z1) + sizeof(sh.z2)) / 4);
    for (int i = tid; i < zw; i += 256) w[i] = 0u;
  }
  __syncthreads();
  // initial hidden state into buffer 0
  for (int i = tid; i < 16 * kUnits; i += 256) {
    const int m = i >> 8, u = i & 255;
    const _Float16 v = (_Float16)p.h0[(size_t)(brow + m) * kUnits + u];
    if (u < kH0)            sh.z0[0][m][64 + u] = v;
    else if (u < kH0 + kH1) sh.z1[0][m][kH0 + (u - kH0)] = v;
    else                    sh.z2[0][m][kH1 + (u - kH0 - kH1)] = v;
  }
  // biases -> LDS (padded with zeros)
  for (int i = tid; i < 3 * 4 * 128; i += 256) {
    const int l = i >> 9, g = (i >> 7) & 3, n = i & 127;
    const int hl = (l == 0) ? kH0 : (l == 1) ? kH1 : kH2;
    sh.bias[l][g][n] = (n < hl) ? p.bias[l][g][n] : 0.f;
  }
  __syncthreads();

  const _Float16* ws0 = p.ws + (size_t)BASE0 * FRAG;
  const _Float16* ws1 = p.ws + (size_t)BASE1 * FRAG;
  const _Float16* ws2 = p.ws + (size_t)BASE2 * FRAG;

  for (int t = 0; t < kT; ++t) {
    const int cb = t & 1, nb = cb ^ 1;
    const bool last = (t == kT - 1);

    { // stage x_t (16 x 64 f32 -> f16) into z0[cb] cols 0..63; 1 float4 per thread
      const int m = tid >> 4, f4 = (tid & 15) << 2;
      const v4f xv = *(const v4f*)(p.x + ((size_t)(brow + m) * kT + t) * kF + f4);
      union { _Float16 h[4]; unsigned long long u64; } pk;
      pk.h[0] = (_Float16)xv.x; pk.h[1] = (_Float16)xv.y;
      pk.h[2] = (_Float16)xv.z; pk.h[3] = (_Float16)xv.w;
      *(unsigned long long*)&sh.z0[cb][m][f4] = pk.u64;
    }
    __syncthreads();

    if (wave < NT0) {  // layer 0: waves 0..4, one N-tile each, all 4 gates
      layer_compute<KT0>(&sh.z0[cb][0][0], 160, ws0, NT0 * KT0 * FRAG, wave, lane,
                         &sh.bias[0][0][0], kH0,
        [&](int mo, int n, float v) {
          sh.z0[nb][mo][64 + n] = (_Float16)v;          // own z, next step
          sh.z1[cb][mo][n]      = (_Float16)v;          // feeds layer 1 this step
          if (last) p.hx[(size_t)(brow + mo) * kUnits + n] = v;
        });
    }
    __syncthreads();

    if (wave < NT1) {  // layer 1: waves 0..3
      layer_compute<KT1>(&sh.z1[cb][0][0], 128, ws1, NT1 * KT1 * FRAG, wave, lane,
                         &sh.bias[1][0][0], kH1,
        [&](int mo, int n, float v) {
          sh.z1[nb][mo][kH0 + n] = (_Float16)v;
          sh.z2[cb][mo][n]       = (_Float16)v;
          if (last) p.hx[(size_t)(brow + mo) * kUnits + kH0 + n] = v;
        });
    }
    __syncthreads();

    {  // layer 2: all 8 waves
      layer_compute<KT2>(&sh.z2[cb][0][0], 192, ws2, NT2 * KT2 * FRAG, wave, lane,
                         &sh.bias[2][0][0], kH2,
        [&](int mo, int n, float v) {
          sh.z2[nb][mo][kH1 + n] = (_Float16)v;
          p.out[((size_t)(brow + mo) * kT + t) * kH2 + n] = v;  // return_sequences
          if (last) p.hx[(size_t)(brow + mo) * kUnits + kH0 + kH1 + n] = v;
        });
    }
    __syncthreads();
  }
}

// ---------------- launch ------------------------------------------------------
extern "C" void kernel_launch(void* const* d_in, const int* in_sizes, int n_in,
                              void* d_out, int out_size, void* d_ws, size_t ws_size,
                              hipStream_t stream) {
  (void)in_sizes; (void)n_in; (void)out_size; (void)ws_size;
  // d_in dict order: x, h0, then per layer l: m, W1, W2, Wa, Wb, b1, b2, ba, bb
  PrepParams pp;
  for (int l = 0; l < 3; ++l) {
    const int base = 2 + 9 * l;
    pp.mask[l] = (const float*)d_in[base];
    for (int g = 0; g < 4; ++g) pp.W[l][g] = (const float*)d_in[base + 1 + g];
  }
  pp.ws = (_Float16*)d_ws;
  hipLaunchKernelGGL(cfc_prep_weights, dim3(NFRAG), dim3(512), 0, stream, pp);

  MainParams mp;
  mp.x  = (const float*)d_in[0];
  mp.h0 = (const float*)d_in[1];
  for (int l = 0; l < 3; ++l) {
    const int base = 2 + 9 * l;
    for (int g = 0; g < 4; ++g) mp.bias[l][g] = (const float*)d_in[base + 5 + g];
  }
  mp.ws  = (const _Float16*)d_ws;
  mp.out = (float*)d_out;
  mp.hx  = mp.out + (size_t)kB * kT * kH2;   // outputs concatenated: (B,T,128) then (B,UNITS)
  hipLaunchKernelGGL(cfc_persistent, dim3(kB / 16), dim3(256), 0, stream, mp);
}
